// MotionCompensation_16810501996875
// MI455X (gfx1250) — matrix operand order repaired and checked by reading, not compiled, not run on hardware
//
#include <hip/hip_runtime.h>
#include <stdint.h>

typedef __attribute__((ext_vector_type(16))) __bf16 v16bf;
typedef __attribute__((ext_vector_type(8)))  __bf16 v8bf;
typedef __attribute__((ext_vector_type(8)))  float  v8f;
typedef unsigned short ushort_t;

// Problem constants
#define Bn 4
#define Cn 64
#define Hn 128
#define Wn 128
#define Gg 8
#define CPGc 8
#define Nk 9
#define Hp 130           // halo-padded spatial dims
#define Wp 130

// Workspace layout (bytes)
#define OFF_TGT  ((size_t)0)           // target f32 NCHW:            16,777,216 B
#define OFF_CAT  ((size_t)16777216)    // cat bf16 NHWC padded 128ch: 17,305,600 B
#define OFF_Y1   ((size_t)34082816)    // y1 bf16 NHWC padded 64ch:    8,652,800 B
#define OFF_W1P  ((size_t)42735616)    // packed w1 fragments:           147,456 B
#define OFF_W2P  ((size_t)42883072)    // packed w2 fragments:            73,728 B
#define ZERO_BYTES ((size_t)(17305600 + 8652800))   // cat + y1 (contiguous)

__device__ __forceinline__ ushort_t f2bf(float f) {
  unsigned int u = __float_as_uint(f);
  unsigned int r = (u + 0x7FFFu + ((u >> 16) & 1u)) >> 16; // round-to-nearest-even
  return (ushort_t)r;
}

// Build a 16-element bf16 fragment from two 16-byte halves.
__device__ __forceinline__ v16bf ld2(const ushort_t* __restrict__ plo,
                                     const ushort_t* __restrict__ phi) {
  v8bf lo = *reinterpret_cast<const v8bf*>(plo);
  v8bf hi = *reinterpret_cast<const v8bf*>(phi);
  v16bf r;
#pragma unroll
  for (int i = 0; i < 8; ++i) { r[i] = lo[i]; r[8 + i] = hi[i]; }
  return r;
}

// Zero the padded activation buffers (borders must be 0 for halo loads)
__global__ void zero_ws(uint4* __restrict__ p, int n16) {
  int t = blockIdx.x * blockDim.x + threadIdx.x;
  if (t < n16) p[t] = make_uint4(0u, 0u, 0u, 0u);
}

// ---------------------------------------------------------------------------
// Pack conv weights (OIHW f32) into WMMA 32x16 bf16 B-fragments.
// Fragment f = ((kk*CC + cc)*4 + nt): the 4 N-tiles of one K-chunk are
// contiguous. Within fragment: lane-major, 16 bf16/lane.
// B layout (16-bit, 32x16): N=lane%16; K = 8*(lane/16) + e for e<8, 16 + 8*(lane/16) + (e-8) for e>=8.
// ---------------------------------------------------------------------------
__global__ void pack_wB(const float* __restrict__ w, ushort_t* __restrict__ wp,
                        int CIN, int total) {
  int t = blockIdx.x * blockDim.x + threadIdx.x;
  if (t >= total) return;
  int CC    = CIN / 32;
  int e     = t & 15;
  int l     = (t >> 4) & 31;
  int rest  = t >> 9;
  int nt    = rest & 3;
  int rest2 = rest >> 2;
  int cc    = rest2 % CC;
  int kk    = rest2 / CC;
  int k  = ((e >> 3) << 4) + ((l >> 4) << 3) + (e & 7);
  int o  = nt * 16 + (l & 15);
  int c  = cc * 32 + k;
  int ky = kk / 3, kx = kk % 3;
  wp[t] = f2bf(w[((size_t)(o * CIN + c) * 3 + ky) * 3 + kx]);
}

// anchor (NCHW f32) -> padded cat channels [64,128) (NHWC bf16)
__global__ void pack_anchor(const float* __restrict__ anchor,
                            ushort_t* __restrict__ cat) {
  int t = blockIdx.x * blockDim.x + threadIdx.x;
  if (t >= Bn * Cn * Hn * Wn) return;
  int x = t & 127, y = (t >> 7) & 127, c = (t >> 14) & 63, b = t >> 20;
  cat[((size_t)(b * Hp + y + 1) * Wp + (x + 1)) * 128 + 64 + c] = f2bf(anchor[t]);
}

// ---------------------------------------------------------------------------
// Deformable stage: one thread per (b, g, y, x). Group weights (pw/mw/cw,
// ~10KB, block-uniform) staged in LDS -> broadcast ds_loads.
// ---------------------------------------------------------------------------
__global__ void deform_kernel(const float* __restrict__ o_in, const float* __restrict__ anchor,
                              const float* __restrict__ pw, const float* __restrict__ pb,
                              const float* __restrict__ mw, const float* __restrict__ mb,
                              const float* __restrict__ cw,
                              float* __restrict__ tgt, ushort_t* __restrict__ cat) {
  __shared__ float spw[18 * 72];
  __shared__ float smw[9 * 72];
  __shared__ float scw[576];

  const int tid = threadIdx.x;
  const int t = blockIdx.x * 256 + tid;
  const int x = t & 127, y = (t >> 7) & 127, g = (t >> 14) & 7, b = t >> 17;

  for (int i = tid; i < 18 * 72; i += 256) spw[i] = pw[(size_t)g * 1296 + i];
  for (int i = tid; i < 9 * 72;  i += 256) smw[i] = mw[(size_t)g * 648 + i];
  for (int i = tid; i < 576;     i += 256) scw[i] = cw[(size_t)g * 576 + i];
  __syncthreads();

  // 3x3 zero-padded patch of the motion slice f (8 channels)
  float fv[CPGc * Nk];
  for (int ci = 0; ci < CPGc; ++ci) {
    const float* fp = o_in + (size_t)(b * Cn + g * CPGc + ci) * Hn * Wn;
    for (int ky = 0; ky < 3; ++ky)
      for (int kx = 0; kx < 3; ++kx) {
        int yy = y + ky - 1, xx = x + kx - 1;
        float v = 0.f;
        if (yy >= 0 && yy < Hn && xx >= 0 && xx < Wn) v = fp[yy * Wn + xx];
        fv[ci * 9 + ky * 3 + kx] = v;
      }
  }

  float off[2 * Nk];
  for (int j = 0; j < 2 * Nk; ++j) {
    float s = pb[g * 18 + j];
    const float* wj = spw + j * 72;
    for (int q = 0; q < 72; ++q) s += wj[q] * fv[q];
    off[j] = s;
  }
  float mm[Nk];
  for (int n = 0; n < Nk; ++n) {
    float s = mb[g * 9 + n];
    const float* wj = smw + n * 72;
    for (int q = 0; q < 72; ++q) s += wj[q] * fv[q];
    mm[n] = 1.f / (1.f + __expf(-s));
  }

  float acc[CPGc];
  for (int co = 0; co < CPGc; ++co) acc[co] = 0.f;

  for (int n = 0; n < Nk; ++n) {
    float pxv = (float)(y + (n / 3)) + off[n];
    float pyv = (float)(x + (n % 3)) + off[9 + n];
    float fx = floorf(pxv), fy = floorf(pyv);
    float qxl = fminf(fmaxf(fx,       0.f), 127.f);
    float qxr = fminf(fmaxf(fx + 1.f, 0.f), 127.f);
    float qyl = fminf(fmaxf(fy,       0.f), 127.f);
    float qyr = fminf(fmaxf(fy + 1.f, 0.f), 127.f);
    float pxc = fminf(fmaxf(pxv, 0.f), 127.f);
    float pyc = fminf(fmaxf(pyv, 0.f), 127.f);
    float glt = (1.f + (qxl - pxc)) * (1.f + (qyl - pyc));
    float grb = (1.f - (qxr - pxc)) * (1.f - (qyr - pyc));
    float glb = (1.f + (qxl - pxc)) * (1.f - (qyr - pyc));
    float grt = (1.f - (qxr - pxc)) * (1.f + (qyl - pyc));
    int ixl = (int)qxl, ixr = (int)qxr, iyl = (int)qyl, iyr = (int)qyr;

    for (int ci = 0; ci < CPGc; ++ci) {
      const float* ap = anchor + (size_t)(b * Cn + g * CPGc + ci) * Hn * Wn;
      float v = glt * ap[ixl * Wn + iyl] + grb * ap[ixr * Wn + iyr]
              + glb * ap[ixl * Wn + iyr] + grt * ap[ixr * Wn + iyl];
      v *= mm[n];
      const float* cwp = scw + ci * 9 + n;     // co stride = 72
      for (int co = 0; co < CPGc; ++co) acc[co] += cwp[co * 72] * v;
    }
  }

  for (int co = 0; co < CPGc; ++co) {
    int tc = g * CPGc + co;
    tgt[((size_t)(b * Cn + tc) * Hn + y) * Wn + x] = acc[co];
    cat[((size_t)(b * Hp + y + 1) * Wp + (x + 1)) * 128 + tc] = f2bf(acc[co]);
  }
}

// ---------------------------------------------------------------------------
// Implicit-GEMM 3x3 conv core: one wave computes 32 pixels x 64 out-channels.
// Per K-chunk: 2 A fragments + 4 B fragments feed 8 WMMAs (1.5 vmem/wmma);
// each B register fragment is reused by 2 WMMAs, each A by 4.
// ---------------------------------------------------------------------------
template <int CIN>
__device__ __forceinline__ void conv_tile_acc8(const ushort_t* __restrict__ xin,
                                               const ushort_t* __restrict__ wp,
                                               int b, int y, int x0, int lane,
                                               v8f acc0[4], v8f acc1[4]) {
  constexpr int CC = CIN / 32;
  const int m = lane & 15;
  const int half8 = (lane >> 4) * 8;
#pragma unroll
  for (int kk = 0; kk < 9; ++kk) {
    const int ky = kk / 3, kx = kk % 3;
    const ushort_t* pix0 = xin + (size_t)((b * Hp + y + ky) * Wp + x0 + m + kx) * CIN;
    const ushort_t* pix1 = pix0 + 16 * CIN;
#pragma unroll
    for (int cc = 0; cc < CC; ++cc) {
      const int c0 = cc * 32;
      v16bf a0 = ld2(pix0 + c0 + half8, pix0 + c0 + 16 + half8);
      v16bf a1 = ld2(pix1 + c0 + half8, pix1 + c0 + 16 + half8);
      const ushort_t* wbase = wp + (size_t)((kk * CC + cc) * 4) * 512 + lane * 16;
#pragma unroll
      for (int nt = 0; nt < 4; ++nt) {
        v16bf bv = ld2(wbase + nt * 512, wbase + nt * 512 + 8);
        acc0[nt] = __builtin_amdgcn_wmma_f32_16x16x32_bf16(false, a0, false, bv,
                                                           (short)0, acc0[nt], false, false);
        acc1[nt] = __builtin_amdgcn_wmma_f32_16x16x32_bf16(false, a1, false, bv,
                                                           (short)0, acc1[nt], false, false);
      }
    }
  }
}

// conv1: cat(128ch padded NHWC bf16) -> y1(64ch padded NHWC bf16), + b1
__global__ void conv1_kernel(const ushort_t* __restrict__ cat,
                             const ushort_t* __restrict__ w1p,
                             const float* __restrict__ b1,
                             ushort_t* __restrict__ y1) {
  const int lane = threadIdx.x & 31;
  const int wid  = threadIdx.x >> 5;           // 4 waves = 4 x-tiles (32px) of one row
  const int blk  = blockIdx.x;
  const int x0 = wid * 32;
  const int y  = blk & 127;
  const int b  = blk >> 7;
  v8f acc0[4], acc1[4];
#pragma unroll
  for (int nt = 0; nt < 4; ++nt)
#pragma unroll
    for (int i = 0; i < 8; ++i) { acc0[nt][i] = 0.f; acc1[nt][i] = 0.f; }
  conv_tile_acc8<128>(cat, w1p, b, y, x0, lane, acc0, acc1);
  const int mbase = (lane >> 4) * 8;           // D: VGPR r -> M=r (lanes 0-15) / 8+r
#pragma unroll
  for (int nt = 0; nt < 4; ++nt) {
    const int o = nt * 16 + (lane & 15);
    const float bv = b1[o];
#pragma unroll
    for (int r = 0; r < 8; ++r) {
      const int M = r + mbase;
      const size_t row = (size_t)(b * Hp + y + 1) * Wp;
      y1[(row + x0 + M + 1) * 64 + o]      = f2bf(acc0[nt][r] + bv);
      y1[(row + x0 + 16 + M + 1) * 64 + o] = f2bf(acc1[nt][r] + bv);
    }
  }
}

// conv2: y1(64ch padded NHWC bf16) -> d_out (f32 NCHW) with +b2 and +target
__global__ void conv2_kernel(const ushort_t* __restrict__ y1,
                             const ushort_t* __restrict__ w2p,
                             const float* __restrict__ b2,
                             const float* __restrict__ tgt,
                             float* __restrict__ out) {
  const int lane = threadIdx.x & 31;
  const int wid  = threadIdx.x >> 5;
  const int blk  = blockIdx.x;
  const int x0 = wid * 32;
  const int y  = blk & 127;
  const int b  = blk >> 7;
  v8f acc0[4], acc1[4];
#pragma unroll
  for (int nt = 0; nt < 4; ++nt)
#pragma unroll
    for (int i = 0; i < 8; ++i) { acc0[nt][i] = 0.f; acc1[nt][i] = 0.f; }
  conv_tile_acc8<64>(y1, w2p, b, y, x0, lane, acc0, acc1);
  const int mbase = (lane >> 4) * 8;
#pragma unroll
  for (int nt = 0; nt < 4; ++nt) {
    const int o = nt * 16 + (lane & 15);
    const float bv = b2[o];
#pragma unroll
    for (int r = 0; r < 8; ++r) {
      const int M = r + mbase;
      const size_t orow = ((size_t)(b * Cn + o) * Hn + y) * Wn;
      const size_t oi0 = orow + x0 + M;
      const size_t oi1 = orow + x0 + 16 + M;
      out[oi0] = tgt[oi0] + acc0[nt][r] + bv;
      out[oi1] = tgt[oi1] + acc1[nt][r] + bv;
    }
  }
}

extern "C" void kernel_launch(void* const* d_in, const int* in_sizes, int n_in,
                              void* d_out, int out_size, void* d_ws, size_t ws_size,
                              hipStream_t stream) {
  const float* o_in   = (const float*)d_in[0];
  const float* anchor = (const float*)d_in[1];
  const float* pw = (const float*)d_in[2];
  const float* pb = (const float*)d_in[3];
  const float* mw = (const float*)d_in[4];
  const float* mb = (const float*)d_in[5];
  const float* cw = (const float*)d_in[6];
  const float* w1 = (const float*)d_in[7];
  const float* b1 = (const float*)d_in[8];
  const float* w2 = (const float*)d_in[9];
  const float* b2 = (const float*)d_in[10];
  float* out = (float*)d_out;

  char* ws = (char*)d_ws;
  float*    tgt = (float*)(ws + OFF_TGT);
  ushort_t* cat = (ushort_t*)(ws + OFF_CAT);
  ushort_t* y1  = (ushort_t*)(ws + OFF_Y1);
  ushort_t* w1p = (ushort_t*)(ws + OFF_W1P);
  ushort_t* w2p = (ushort_t*)(ws + OFF_W2P);

  // zero padded activation buffers (cat + y1 are contiguous in ws)
  {
    int n16 = (int)(ZERO_BYTES / 16);
    zero_ws<<<(n16 + 255) / 256, 256, 0, stream>>>((uint4*)(ws + OFF_CAT), n16);
  }
  // weight fragment packing
  pack_wB<<<(73728 + 255) / 256, 256, 0, stream>>>(w1, w1p, 128, 73728);
  pack_wB<<<(36864 + 255) / 256, 256, 0, stream>>>(w2, w2p, 64, 36864);
  // anchor -> cat[64:128)
  pack_anchor<<<(Bn * Cn * Hn * Wn) / 256, 256, 0, stream>>>(anchor, cat);
  // deformable stage -> target + cat[0:64)
  deform_kernel<<<(Bn * Gg * Hn * Wn) / 256, 256, 0, stream>>>(
      o_in, anchor, pw, pb, mw, mb, cw, tgt, cat);
  // WMMA convs: 512 blocks x 4 waves, each wave = 32px x 64 outch
  conv1_kernel<<<Hn * Bn, 128, 0, stream>>>(cat, w1p, b1, y1);
  conv2_kernel<<<Hn * Bn, 128, 0, stream>>>(y1, w2p, b2, tgt, out);
}
